// MoELinearA_14070312862079
// MI455X (gfx1250) — matrix operand order repaired and checked by reading, not compiled, not run on hardware
//
#include <hip/hip_runtime.h>

typedef __attribute__((ext_vector_type(16))) __bf16 v16bf;
typedef __attribute__((ext_vector_type(8)))  float  v8f;
typedef __attribute__((ext_vector_type(4)))  int    v4i;

union AF {
  unsigned short s[16];
  uint4 u[2];
  v16bf v;
};

static constexpr int kD = 4096;     // in_features
static constexpr int kR = 64;       // rank per expert
static constexpr int kE = 8;        // experts
static constexpr int kM = 8192;     // B*S tokens
static constexpr int LDSS = 40;     // LDS row stride (bf16 elems), padded for bank spread

#ifndef __has_builtin
#define __has_builtin(x) 0
#endif
#if __has_builtin(__builtin_amdgcn_global_load_async_to_lds_b128) && \
    __has_builtin(__builtin_amdgcn_s_wait_asynccnt)
#define USE_ASYNC 1
#define AS_G __attribute__((address_space(1)))
#define AS_L __attribute__((address_space(3)))
#else
#define USE_ASYNC 0
#endif

__device__ __forceinline__ unsigned int bf16_rne(float f) {
  unsigned int u = __float_as_uint(f);
  u += 0x7FFFu + ((u >> 16) & 1u);   // round-to-nearest-even to bf16
  return u >> 16;
}
__device__ __forceinline__ float bf16_f32(unsigned int h) {
  return __uint_as_float(h << 16);
}

// ---------- pre-pass: split fp32 -> bf16 hi/lo planes (memory-bound, runs once) ----------
__global__ __launch_bounds__(256)
void split_bf16(const float* __restrict__ src,
                unsigned short* __restrict__ hi,
                unsigned short* __restrict__ lo) {
  const size_t base = ((size_t)blockIdx.x * 256 + threadIdx.x) * 8;
  float4 a = *(const float4*)(src + base);
  float4 b = *(const float4*)(src + base + 4);
  float v[8] = {a.x, a.y, a.z, a.w, b.x, b.y, b.z, b.w};
  unsigned int h[8], l[8];
#pragma unroll
  for (int i = 0; i < 8; ++i) {
    h[i] = bf16_rne(v[i]);
    l[i] = bf16_rne(v[i] - bf16_f32(h[i]));
  }
  *(uint4*)(hi + base) = make_uint4(h[0] | (h[1] << 16), h[2] | (h[3] << 16),
                                    h[4] | (h[5] << 16), h[6] | (h[7] << 16));
  *(uint4*)(lo + base) = make_uint4(l[0] | (l[1] << 16), l[2] | (l[3] << 16),
                                    l[4] | (l[5] << 16), l[6] | (l[7] << 16));
}

#if USE_ASYNC
__device__ __forceinline__ void async_b128(const unsigned short* g, unsigned short* l) {
  __builtin_amdgcn_global_load_async_to_lds_b128(
      (AS_G v4i*)(AS_G void*)(void*)g,
      (AS_L v4i*)(AS_L void*)(void*)l, 0, 0);
}
#endif

// ---------- GEMM: y[e, m, r] = sum_k x[m,k] * W[e,r,k], split-precision bf16x2 ----------
__global__ __launch_bounds__(256, 1)
void moe_gemm_bf16x2(const unsigned short* __restrict__ xhi,
                     const unsigned short* __restrict__ xlo,
                     const unsigned short* __restrict__ whi,
                     const unsigned short* __restrict__ wlo,
                     float* __restrict__ y) {
  __shared__ alignas(16) unsigned short sh[2][kR * LDSS];   // W chunk, hi plane (double buffered)
  __shared__ alignas(16) unsigned short sl[2][kR * LDSS];   // W chunk, lo plane

  const int tid  = threadIdx.x;
  const int wave = tid >> 5;
  const int lane = tid & 31;
  const int lr   = lane & 15;   // A row / B column within fragment
  const int g    = lane >> 4;   // half-wave group

  const int e    = blockIdx.y;
  const int m0   = blockIdx.x * 128;
  const int arow = m0 + wave * 16 + lr;

  const unsigned short* xh = xhi + (size_t)arow * kD;
  const unsigned short* xl = xlo + (size_t)arow * kD;

  // cooperative W staging: thread -> row tid>>2 (0..63), 8 contiguous bf16 (16B)
  const int wr = tid >> 2;
  const int wc = (tid & 3) * 8;
  const unsigned short* gwh = whi + ((size_t)e * kR + wr) * kD + wc;
  const unsigned short* gwl = wlo + ((size_t)e * kR + wr) * kD + wc;
  const int sidx = wr * LDSS + wc;

  v8f acc[4] = {};

  // prologue: stage chunk 0 into buffer 0
#if USE_ASYNC
  async_b128(gwh, &sh[0][sidx]);
  async_b128(gwl, &sl[0][sidx]);
  __builtin_amdgcn_s_wait_asynccnt(0);
#else
  *(uint4*)&sh[0][sidx] = *(const uint4*)gwh;
  *(uint4*)&sl[0][sidx] = *(const uint4*)gwl;
#endif
  __syncthreads();

  for (int kc = 0; kc < kD; kc += 32) {
    const int cur = (kc >> 5) & 1;
    const int nxt = cur ^ 1;
    const bool more = (kc + 32) < kD;

    // prefetch next W chunk into the other buffer (overlaps with compute below)
#if USE_ASYNC
    if (more) {
      async_b128(gwh + kc + 32, &sh[nxt][sidx]);
      async_b128(gwl + kc + 32, &sl[nxt][sidx]);
    }
#else
    uint4 pw_h = {}, pw_l = {};
    if (more) {
      pw_h = *(const uint4*)(gwh + kc + 32);
      pw_l = *(const uint4*)(gwl + kc + 32);
    }
#endif

    // A fragments: contiguous k order (same as B), loaded straight from bf16 planes.
    // elem i -> K = kc + 16g + i
    AF ah, al;
    const int kb = kc + 16 * g;
    ah.u[0] = *(const uint4*)(xh + kb);
    ah.u[1] = *(const uint4*)(xh + kb + 8);
    al.u[0] = *(const uint4*)(xl + kb);
    al.u[1] = *(const uint4*)(xl + kb + 8);

    // B fragments from LDS (lane = column, elem i -> K = kc + 16g + i);
    // 3-term split-precision accumulate per N-tile
#pragma unroll
    for (int t = 0; t < 4; ++t) {
      const int bo = (t * 16 + lr) * LDSS + 16 * g;
      AF bh, bl;
      bh.u[0] = *(const uint4*)&sh[cur][bo];
      bh.u[1] = *(const uint4*)&sh[cur][bo + 8];
      bl.u[0] = *(const uint4*)&sl[cur][bo];
      bl.u[1] = *(const uint4*)&sl[cur][bo + 8];
      acc[t] = __builtin_amdgcn_wmma_f32_16x16x32_bf16(
          false, ah.v, false, bh.v, (short)0, acc[t], false, false);
      acc[t] = __builtin_amdgcn_wmma_f32_16x16x32_bf16(
          false, ah.v, false, bl.v, (short)0, acc[t], false, false);
      acc[t] = __builtin_amdgcn_wmma_f32_16x16x32_bf16(
          false, al.v, false, bh.v, (short)0, acc[t], false, false);
    }

#if USE_ASYNC
    __builtin_amdgcn_s_wait_asynccnt(0);
#else
    if (more) {
      *(uint4*)&sh[nxt][sidx] = pw_h;
      *(uint4*)&sl[nxt][sidx] = pw_l;
    }
#endif
    __syncthreads();
  }

  // epilogue: C/D layout (VGPR j -> M = j + 8g, N = lr) -> y[e][token][r]
  float* ye = y + (size_t)e * kM * kR;
  const int mbase = m0 + wave * 16 + 8 * g;
#pragma unroll
  for (int t = 0; t < 4; ++t) {
    const int col = t * 16 + lr;
#pragma unroll
    for (int j = 0; j < 8; ++j) {
      ye[(size_t)(mbase + j) * kR + col] = acc[t][j];
    }
  }
}

extern "C" void kernel_launch(void* const* d_in, const int* in_sizes, int n_in,
                              void* d_out, int out_size, void* d_ws, size_t ws_size,
                              hipStream_t stream) {
  const float* x = (const float*)d_in[0];   // [4,2048,4096] fp32
  const float* W = (const float*)d_in[1];   // [8,64,4096]  fp32
  float* y = (float*)d_out;                 // [8,8192,64]  fp32

  // workspace layout: bf16 hi/lo planes for x and W (~143 MB)
  unsigned short* xhi = (unsigned short*)d_ws;
  unsigned short* xlo = xhi + (size_t)kM * kD;
  unsigned short* whi = xlo + (size_t)kM * kD;
  unsigned short* wlo = whi + (size_t)kE * kR * kD;

  const int xElems = kM * kD;            // 33,554,432
  const int wElems = kE * kR * kD;       //  2,097,152

  hipLaunchKernelGGL(split_bf16, dim3(xElems / 2048), dim3(256), 0, stream, x, xhi, xlo);
  hipLaunchKernelGGL(split_bf16, dim3(wElems / 2048), dim3(256), 0, stream, W, whi, wlo);

  hipLaunchKernelGGL(moe_gemm_bf16x2, dim3(kM / 128, kE, 1), dim3(256, 1, 1), 0, stream,
                     xhi, xlo, whi, wlo, y);
}
